// TransformerBlock_70909910057653
// MI455X (gfx1250) — matrix operand order repaired and checked by reading, not compiled, not run on hardware
//
#include <hip/hip_runtime.h>
#include <hip/hip_bf16.h>

typedef __bf16 bf16_t;
typedef __attribute__((ext_vector_type(16))) __bf16 v16bf;
typedef __attribute__((ext_vector_type(8)))  __bf16 v8bf;
typedef __attribute__((ext_vector_type(8)))  float  v8f;

#define BATCH 4
#define CCH 192
#define HH 256
#define WW 256
#define NPIX 65536         // HH*WW
#define HEADS 8
#define CHH 24             // channels per head
#define HID 384            // ffn hidden

__device__ __forceinline__ bf16_t f2bf(float f) {
  unsigned u = __builtin_bit_cast(unsigned, f);
  unsigned r = (u + 0x7FFFu + ((u >> 16) & 1u)) >> 16;
  unsigned short s = (unsigned short)r;
  return __builtin_bit_cast(bf16_t, s);
}

// ---------------------------------------------------------------- convert f32 -> bf16
__global__ __launch_bounds__(256)
void cvt_bf16_kernel(const float* __restrict__ in, bf16_t* __restrict__ out, long n) {
  long i = (long)blockIdx.x * 256 + threadIdx.x;
  if (i < n) out[i] = f2bf(in[i]);
}

// ---------------------------------------------------------------- LayerNorm over C per pixel -> bf16
__global__ __launch_bounds__(256)
void ln_c_kernel(const float* __restrict__ X, const float* __restrict__ w,
                 const float* __restrict__ bias, bf16_t* __restrict__ Out) {
  long p = (long)blockIdx.x * 256 + threadIdx.x;       // over BATCH*NPIX
  long b = p / NPIX, hw = p % NPIX;
  const float* xb = X + (b * CCH) * (long)NPIX + hw;
  float s = 0.f, s2 = 0.f;
  for (int c = 0; c < CCH; ++c) { float v = xb[(long)c * NPIX]; s += v; s2 += v * v; }
  float mu = s * (1.0f / CCH);
  float var = s2 * (1.0f / CCH) - mu * mu;
  float rstd = rsqrtf(var + 1e-5f);
  bf16_t* ob = Out + (b * CCH) * (long)NPIX + hw;
  for (int c = 0; c < CCH; ++c) {
    float v = xb[(long)c * NPIX];
    ob[(long)c * NPIX] = f2bf((v - mu) * rstd * w[c] + bias[c]);
  }
}

// ---------------------------------------------------------------- bf16 WMMA GEMM: Y[b] = W(MxK) * X[b](KxN) (+Res)
// block tile 64(M) x 128(N), K-step 32, 8 waves each computing 32x32.
// LDS B tile stored n-major as packed k-pair dwords with 16B-chunk XOR swizzle
// so every fragment load is ds_load_b128. Double-buffered, register-staged.
__global__ __launch_bounds__(256)
void gemm_bf16_wmma(const bf16_t* __restrict__ Wm,   // [M,K] bf16
                    const bf16_t* __restrict__ Xm,   // [B][K][N] bf16
                    float* __restrict__ Ym,          // [B][M][N] f32
                    const float* __restrict__ Res,   // same shape as Y, may be null
                    int M, int K, long strideXb, long strideYb) {
  __shared__ __align__(16) bf16_t  As[2][64][32];     // 8 KB
  __shared__ __align__(16) unsigned Bt[2][128 * 16];  // 16 KB, [n][kpair dword] swizzled
  const int N = NPIX;
  int b  = blockIdx.z;
  int m0 = blockIdx.y * 64;
  int n0 = blockIdx.x * 128;
  const bf16_t* Xb = Xm + (long)b * strideXb;
  float* Yb = Ym + (long)b * strideYb;

  int tid  = threadIdx.x;
  int wave = tid >> 5, lane = tid & 31;
  int wm = wave >> 2, wn = wave & 3;       // 2 x 4 wave grid (32M x 32N per wave)
  int hl = lane >> 4, mr = lane & 15;

  // cooperative A load: row tid>>2, 8 bf16 at (tid&3)*8
  int arow = tid >> 2, ach = (tid & 3) << 3;
  // cooperative B load: k-pair tid>>4 (rows 2kp,2kp+1), n block (tid&15)*8
  int kp  = tid >> 4;
  int bn0 = (tid & 15) << 3;
  int physd = (((kp >> 2) ^ ((bn0 >> 3) & 3)) << 2) | (kp & 3);   // swizzled dword-in-row

  v8f acc[2][2];
  #pragma unroll
  for (int i = 0; i < 2; ++i)
    #pragma unroll
    for (int t = 0; t < 2; ++t)
      #pragma unroll
      for (int r = 0; r < 8; ++r) acc[i][t][r] = 0.0f;

  const int nt = K / 32;
  uint4 ra, rb0, rb1;
  // prologue: tile 0 -> regs -> LDS buf 0
  ra  = *(const uint4*)(Wm + (long)(m0 + arow) * K + ach);
  rb0 = *(const uint4*)(Xb + (long)(2 * kp)     * N + n0 + bn0);
  rb1 = *(const uint4*)(Xb + (long)(2 * kp + 1) * N + n0 + bn0);
  {
    *(uint4*)(&As[0][arow][ach]) = ra;
    const unsigned short* sa = (const unsigned short*)&rb0;
    const unsigned short* sb = (const unsigned short*)&rb1;
    #pragma unroll
    for (int j = 0; j < 8; ++j)
      Bt[0][(bn0 + j) * 16 + physd] = (unsigned)sa[j] | ((unsigned)sb[j] << 16);
  }
  __syncthreads();

  for (int kt = 0; kt < nt; ++kt) {
    int cur = kt & 1, nxt = cur ^ 1;
    if (kt + 1 < nt) {
      int k0 = (kt + 1) * 32;
      ra  = *(const uint4*)(Wm + (long)(m0 + arow) * K + k0 + ach);
      rb0 = *(const uint4*)(Xb + (long)(k0 + 2 * kp)     * N + n0 + bn0);
      rb1 = *(const uint4*)(Xb + (long)(k0 + 2 * kp + 1) * N + n0 + bn0);
      if (kt + 2 < nt) {
        __builtin_prefetch(Wm + (long)(m0 + arow) * K + k0 + 32 + ach, 0, 1);
        __builtin_prefetch(Xb + (long)(k0 + 32 + 2 * kp) * N + n0 + bn0, 0, 1);
      }
    }

    // fragments (all ds_load_b128)
    v16bf af[2], bfr[2];
    #pragma unroll
    for (int fr = 0; fr < 2; ++fr) {
      int m = wm * 32 + fr * 16 + mr;
      v8bf lo = *(const v8bf*)(&As[cur][m][hl * 16]);
      v8bf hi = *(const v8bf*)(&As[cur][m][hl * 16 + 8]);
      af[fr] = __builtin_shufflevector(lo, hi, 0,1,2,3,4,5,6,7,8,9,10,11,12,13,14,15);
    }
    #pragma unroll
    for (int t = 0; t < 2; ++t) {
      int n = wn * 32 + t * 16 + mr;
      int c3 = (n >> 3) & 3;
      const unsigned* row = &Bt[cur][n * 16];
      v8bf lo = *(const v8bf*)(row + (((hl * 2)     ^ c3) << 2));
      v8bf hi = *(const v8bf*)(row + (((hl * 2 + 1) ^ c3) << 2));
      bfr[t] = __builtin_shufflevector(lo, hi, 0,1,2,3,4,5,6,7,8,9,10,11,12,13,14,15);
    }
    #pragma unroll
    for (int fr = 0; fr < 2; ++fr)
      #pragma unroll
      for (int t = 0; t < 2; ++t)
        acc[fr][t] = __builtin_amdgcn_wmma_f32_16x16x32_bf16(
            false, af[fr], false, bfr[t], (short)0, acc[fr][t], false, false);

    if (kt + 1 < nt) {
      *(uint4*)(&As[nxt][arow][ach]) = ra;
      const unsigned short* sa = (const unsigned short*)&rb0;
      const unsigned short* sb = (const unsigned short*)&rb1;
      #pragma unroll
      for (int j = 0; j < 8; ++j)
        Bt[nxt][(bn0 + j) * 16 + physd] = (unsigned)sa[j] | ((unsigned)sb[j] << 16);
    }
    __syncthreads();
  }

  // epilogue: C layout per 16x16 tile: half -> +0/+8 rows, VGPR r -> row, N = mr
  #pragma unroll
  for (int fr = 0; fr < 2; ++fr) {
    int mb = m0 + wm * 32 + fr * 16 + hl * 8;
    #pragma unroll
    for (int t = 0; t < 2; ++t) {
      int n = n0 + wn * 32 + t * 16 + mr;
      #pragma unroll
      for (int r = 0; r < 8; ++r) {
        long idx = (long)(mb + r) * N + n;
        float v = acc[fr][t][r];
        if (Res) v += Res[(long)b * strideYb + idx];
        Yb[idx] = v;
      }
    }
  }
}

// ---------------------------------------------------------------- depthwise 3x3, pad 1
__global__ __launch_bounds__(256)
void dwconv3x3_kernel(const float* __restrict__ in, const float* __restrict__ wdw,
                      float* __restrict__ out, int CH, long total) {
  long idx = (long)blockIdx.x * 256 + threadIdx.x;
  if (idx >= total) return;
  int w = (int)(idx % WW); long t = idx / WW;
  int h = (int)(t % HH);   t /= HH;
  int c = (int)(t % CH);   long b = t / CH;
  const float* ip = in + ((b * CH + c) * (long)HH) * WW;
  const float* wp = wdw + (long)c * 9;
  float s = 0.f;
  #pragma unroll
  for (int i = -1; i <= 1; ++i)
    #pragma unroll
    for (int j = -1; j <= 1; ++j) {
      int hh = h + i, ww2 = w + j;
      if (hh >= 0 && hh < HH && ww2 >= 0 && ww2 < WW)
        s += ip[(long)hh * WW + ww2] * wp[(i + 1) * 3 + (j + 1)];
    }
  out[idx] = s;
}

// ---------------------------------------------------------------- row L2 norms (global attention)
__global__ __launch_bounds__(256)
void rownorm_global_kernel(const float* __restrict__ qkv, float* __restrict__ norms) {
  __shared__ float red[256];
  int row = blockIdx.x;                      // 0 .. B*384-1
  int b = row / 384, c = row % 384;          // q: 0-191, k: 192-383 (contiguous in qkv)
  const float* base = qkv + ((long)b * 576 + c) * NPIX;
  int tid = threadIdx.x;
  float s = 0.f;
  for (int n = tid; n < NPIX; n += 256) { float v = base[n]; s += v * v; }
  red[tid] = s; __syncthreads();
  for (int st = 128; st > 0; st >>= 1) { if (tid < st) red[tid] += red[tid + st]; __syncthreads(); }
  if (tid == 0) norms[row] = fmaxf(sqrtf(red[0]), 1e-12f);
}

// ---------------------------------------------------------------- row L2 norms per window (local)
__global__ __launch_bounds__(256)
void rownorm_local_kernel(const float* __restrict__ qkv, float* __restrict__ norms) {
  __shared__ float red[256];
  int id = blockIdx.x;                       // (b*384 + c)*16 + win
  int win = id & 15; int bc = id >> 4;
  int c = bc % 384;  int b = bc / 384;
  int wj = win & 3, wi = win >> 2;
  const float* base = qkv + ((long)b * 576 + c) * NPIX;
  long wbase = (long)(wi * 64) * WW + wj * 64;
  int tid = threadIdx.x;
  float s = 0.f;
  for (int e = tid; e < 4096; e += 256) {
    int r = e >> 6, col = e & 63;
    float v = base[wbase + (long)r * WW + col];
    s += v * v;
  }
  red[tid] = s; __syncthreads();
  for (int st = 128; st > 0; st >>= 1) { if (tid < st) red[tid] += red[tid + st]; __syncthreads(); }
  if (tid == 0) norms[id] = fmaxf(sqrtf(red[0]), 1e-12f);
}

// ---------------------------------------------------------------- Gram (global): 24x24 per (b,head)
__global__ __launch_bounds__(576)
void gram_global_kernel(const float* __restrict__ qkv, const float* __restrict__ norms,
                        const float* __restrict__ temp, float* __restrict__ gram) {
  __shared__ float qs[24][128];
  __shared__ float ks[24][128];
  int bh = blockIdx.x; int b = bh / HEADS, h = bh % HEADS;
  int tid = threadIdx.x; int c = tid / 24, d = tid % 24;
  const float* qb = qkv + ((long)b * 576 + h * CHH) * NPIX;
  const float* kb = qkv + ((long)b * 576 + 192 + h * CHH) * NPIX;
  float acc = 0.f;
  for (int n0 = 0; n0 < NPIX; n0 += 128) {
    for (int e = tid; e < 3072; e += 576) {
      int cc = e >> 7, nn = e & 127;
      qs[cc][nn] = qb[(long)cc * NPIX + n0 + nn];
      ks[cc][nn] = kb[(long)cc * NPIX + n0 + nn];
    }
    __syncthreads();
    for (int j = 0; j < 128; ++j) acc += qs[c][j] * ks[d][j];
    __syncthreads();
  }
  float nq = norms[(long)b * 384 + h * CHH + c];
  float nk = norms[(long)b * 384 + 192 + h * CHH + d];
  gram[(long)bh * 576 + c * 24 + d] = acc / (nq * nk) * temp[h];
}

// ---------------------------------------------------------------- Gram (local windows)
__global__ __launch_bounds__(576)
void gram_local_kernel(const float* __restrict__ qkv, const float* __restrict__ norms,
                       const float* __restrict__ temp, float* __restrict__ gram) {
  __shared__ float qs[24][128];
  __shared__ float ks[24][128];
  int bhw = blockIdx.x;                      // ((b*8+h)*16 + win)
  int win = bhw & 15, h = (bhw >> 4) & 7, b = bhw >> 7;
  int wj = win & 3, wi = win >> 2;
  int tid = threadIdx.x; int c = tid / 24, d = tid % 24;
  long qoff = ((long)b * 576 + h * CHH) * NPIX;
  long koff = ((long)b * 576 + 192 + h * CHH) * NPIX;
  long wbase = (long)(wi * 64) * WW + wj * 64;
  float acc = 0.f;
  for (int n0 = 0; n0 < 4096; n0 += 128) {
    for (int e = tid; e < 3072; e += 576) {
      int cc = e >> 7, nn = e & 127;
      int el = n0 + nn; int r = el >> 6, col = el & 63;
      long a = wbase + (long)r * WW + col;
      qs[cc][nn] = qkv[qoff + (long)cc * NPIX + a];
      ks[cc][nn] = qkv[koff + (long)cc * NPIX + a];
    }
    __syncthreads();
    for (int j = 0; j < 128; ++j) acc += qs[c][j] * ks[d][j];
    __syncthreads();
  }
  float nq = norms[((long)b * 384 + h * CHH + c) * 16 + win];
  float nk = norms[((long)b * 384 + 192 + h * CHH + d) * 16 + win];
  gram[(long)bhw * 576 + c * 24 + d] = acc / (nq * nk) * temp[h];
}

// ---------------------------------------------------------------- softmax over 24-wide rows (in place)
__global__ __launch_bounds__(256)
void softmax24_kernel(float* __restrict__ g, int nrows) {
  int r = blockIdx.x * 256 + threadIdx.x;
  if (r >= nrows) return;
  float* row = g + (long)r * 24;
  float m = row[0];
  for (int i = 1; i < 24; ++i) m = fmaxf(m, row[i]);
  float e[24]; float s = 0.f;
  for (int i = 0; i < 24; ++i) { e[i] = expf(row[i] - m); s += e[i]; }
  float inv = 1.f / s;
  for (int i = 0; i < 24; ++i) row[i] = e[i] * inv;
}

// ---------------------------------------------------------------- attn @ V (global) -> bf16 [B,192,N]
__global__ __launch_bounds__(256)
void attnv_global_kernel(const float* __restrict__ qkv, const float* __restrict__ attn,
                         bf16_t* __restrict__ out) {
  __shared__ float a[24][24];
  int bh = blockIdx.y; int b = bh / HEADS, h = bh % HEADS;
  int tid = threadIdx.x;
  for (int e = tid; e < 576; e += 256) a[e / 24][e % 24] = attn[(long)bh * 576 + e];
  __syncthreads();
  int n = blockIdx.x * 256 + tid;
  const float* vb = qkv + ((long)b * 576 + 384 + h * CHH) * NPIX;
  float vv[24];
  for (int d = 0; d < 24; ++d) vv[d] = vb[(long)d * NPIX + n];
  bf16_t* ob = out + ((long)b * CCH + h * CHH) * NPIX;
  for (int c = 0; c < 24; ++c) {
    float s = 0.f;
    for (int d = 0; d < 24; ++d) s += a[c][d] * vv[d];
    ob[(long)c * NPIX + n] = f2bf(s);
  }
}

// ---------------------------------------------------------------- attn @ V (local, * tanh(local_weight)) -> bf16
__global__ __launch_bounds__(256)
void attnv_local_kernel(const float* __restrict__ qkv, const float* __restrict__ attn,
                        const float* __restrict__ lw, bf16_t* __restrict__ out) {
  __shared__ float a[24][24];
  int bhw = blockIdx.y;
  int win = bhw & 15, h = (bhw >> 4) & 7, b = bhw >> 7;
  int wj = win & 3, wi = win >> 2;
  int tid = threadIdx.x;
  for (int e = tid; e < 576; e += 256) a[e / 24][e % 24] = attn[(long)bhw * 576 + e];
  __syncthreads();
  int e = blockIdx.x * 256 + tid;           // 0..4095 within window
  int r = e >> 6, col = e & 63;
  long hw = (long)(wi * 64 + r) * WW + wj * 64 + col;
  const float* vb = qkv + ((long)b * 576 + 384 + h * CHH) * NPIX;
  float vv[24];
  for (int d = 0; d < 24; ++d) vv[d] = vb[(long)d * NPIX + hw];
  float tw = tanhf(lw[wi * 4 + wj]);
  bf16_t* ob = out + ((long)b * CCH + h * CHH) * NPIX;
  for (int c = 0; c < 24; ++c) {
    float s = 0.f;
    for (int d = 0; d < 24; ++d) s += a[c][d] * vv[d];
    ob[(long)c * NPIX + hw] = f2bf(s * tw);
  }
}

// ---------------------------------------------------------------- x + (1-ld)*glo + ld*loc (in place into loc buffer)
__global__ __launch_bounds__(256)
void combine_kernel(const float* __restrict__ x, const float* __restrict__ ld,
                    const float* __restrict__ glo, float* __restrict__ loc_io) {
  long idx = (long)blockIdx.x * 256 + threadIdx.x;     // B*C*N
  long b = idx / ((long)CCH * NPIX);
  long hw = idx % NPIX;
  float l = ld[b * NPIX + hw];
  loc_io[idx] = x[idx] + (1.f - l) * glo[idx] + l * loc_io[idx];
}

// ---------------------------------------------------------------- gelu(x1)*x2 gate -> bf16 [B,384,N]
__global__ __launch_bounds__(256)
void gate_kernel(const float* __restrict__ y, bf16_t* __restrict__ out) {
  long idx = (long)blockIdx.x * 256 + threadIdx.x;     // B*384*N
  long b = idx / ((long)HID * NPIX);
  long rem = idx - b * (long)HID * NPIX;
  long c = rem / NPIX; long hw = rem % NPIX;
  float a = y[(b * 768 + c) * (long)NPIX + hw];
  float g = 0.5f * a * (1.f + erff(a * 0.70710678118654752f));
  float x2 = y[(b * 768 + HID + c) * (long)NPIX + hw];
  out[idx] = f2bf(g * x2);
}

// ================================================================ launcher
extern "C" void kernel_launch(void* const* d_in, const int* in_sizes, int n_in,
                              void* d_out, int out_size, void* d_ws, size_t ws_size,
                              hipStream_t stream) {
  const float* x        = (const float*)d_in[0];
  const float* lightd   = (const float*)d_in[1];
  const float* n1_w     = (const float*)d_in[2];
  const float* n1_b     = (const float*)d_in[3];
  const float* gqkv_w   = (const float*)d_in[4];
  const float* gqkv_dw  = (const float*)d_in[5];
  const float* gproj_w  = (const float*)d_in[6];
  const float* g_temp   = (const float*)d_in[7];
  const float* lqkv_w   = (const float*)d_in[8];
  const float* lqkv_dw  = (const float*)d_in[9];
  const float* lproj_w  = (const float*)d_in[10];
  const float* l_temp   = (const float*)d_in[11];
  const float* local_w  = (const float*)d_in[12];
  const float* n2_w     = (const float*)d_in[13];
  const float* n2_b     = (const float*)d_in[14];
  const float* ffn_in_w = (const float*)d_in[15];
  const float* ffn_dw   = (const float*)d_in[16];
  const float* ffn_outw = (const float*)d_in[17];
  float* out = (float*)d_out;

  char* ws = (char*)d_ws;
  const long N = NPIX;
  // bf16 weight pool (element offsets)
  bf16_t* wpool = (bf16_t*)(ws);
  const long o_wgqkv = 0, o_wlqkv = 110592, o_wgproj = 221184, o_wlproj = 258048;
  const long o_wffn1 = 294912, o_wffn2 = 442368;           // end 516096 elems
  const size_t OFF_A = 1u << 20;
  const size_t SZ_A  = (size_t)BATCH * 768 * N * 4;        // 805 MB
  const size_t OFF_B = OFF_A + SZ_A;
  const size_t OFF_C = OFF_B + SZ_A;                        // 201 MB region
  const size_t SZ_C  = (size_t)BATCH * CCH * N * 4;
  const size_t OFF_D = OFF_C + SZ_C;                        // bf16 act pool (403 MB)
  const size_t SZ_D1 = (size_t)BATCH * CCH * N * 2;         // 100 MB (n1/n2)
  const size_t OFF_S = OFF_D + (size_t)BATCH * 768 * N * 2;

  float*  bufA  = (float*)(ws + OFF_A);
  float*  bufB  = (float*)(ws + OFF_B);
  float*  bufC  = (float*)(ws + OFF_C);
  bf16_t* nrm   = (bf16_t*)(ws + OFF_D);                    // n1 / n2 (bf16)
  bf16_t* actD  = (bf16_t*)(ws + OFF_D + SZ_D1);            // attn outs / gate (bf16)
  float*  gnorm = (float*)(ws + OFF_S);                     // 1536
  float*  lnorm = gnorm + 2048;                             // 24576
  float*  ggram = lnorm + 24576;                            // 18432
  float*  lgram = ggram + 18432;                            // 294912

  // 1. weights -> bf16
  cvt_bf16_kernel<<<(110592 + 255) / 256, 256, 0, stream>>>(gqkv_w,   wpool + o_wgqkv, 110592);
  cvt_bf16_kernel<<<(110592 + 255) / 256, 256, 0, stream>>>(lqkv_w,   wpool + o_wlqkv, 110592);
  cvt_bf16_kernel<<<(36864  + 255) / 256, 256, 0, stream>>>(gproj_w,  wpool + o_wgproj, 36864);
  cvt_bf16_kernel<<<(36864  + 255) / 256, 256, 0, stream>>>(lproj_w,  wpool + o_wlproj, 36864);
  cvt_bf16_kernel<<<(147456 + 255) / 256, 256, 0, stream>>>(ffn_in_w, wpool + o_wffn1, 147456);
  cvt_bf16_kernel<<<(73728  + 255) / 256, 256, 0, stream>>>(ffn_outw, wpool + o_wffn2, 73728);

  // 2. n1 = LN(x)
  ln_c_kernel<<<(BATCH * N) / 256, 256, 0, stream>>>(x, n1_w, n1_b, nrm);

  // 3. global qkv GEMM -> bufA, dwconv -> bufB
  gemm_bf16_wmma<<<dim3(N / 128, 576 / 64, BATCH), 256, 0, stream>>>(
      wpool + o_wgqkv, nrm, bufA, nullptr, 576, 192, (long)CCH * N, (long)576 * N);
  {
    long tot = (long)BATCH * 576 * N;
    dwconv3x3_kernel<<<(unsigned)(tot / 256), 256, 0, stream>>>(bufA, gqkv_dw, bufB, 576, tot);
  }
  // 4. global channel attention
  rownorm_global_kernel<<<BATCH * 384, 256, 0, stream>>>(bufB, gnorm);
  gram_global_kernel<<<BATCH * HEADS, 576, 0, stream>>>(bufB, gnorm, g_temp, ggram);
  softmax24_kernel<<<(BATCH * HEADS * 24 + 255) / 256, 256, 0, stream>>>(ggram, BATCH * HEADS * 24);
  attnv_global_kernel<<<dim3(N / 256, BATCH * HEADS), 256, 0, stream>>>(bufB, ggram, actD);

  // 5. glo = gproj GEMM -> bufA[0:201MB]
  gemm_bf16_wmma<<<dim3(N / 128, CCH / 64, BATCH), 256, 0, stream>>>(
      wpool + o_wgproj, actD, bufA, nullptr, CCH, CCH, (long)CCH * N, (long)CCH * N);

  // 6. local qkv GEMM -> bufA tail, dwconv -> bufB
  float* lpre = (float*)(ws + OFF_A + SZ_C);
  gemm_bf16_wmma<<<dim3(N / 128, 576 / 64, BATCH), 256, 0, stream>>>(
      wpool + o_wlqkv, nrm, lpre, nullptr, 576, 192, (long)CCH * N, (long)576 * N);
  {
    long tot = (long)BATCH * 576 * N;
    dwconv3x3_kernel<<<(unsigned)(tot / 256), 256, 0, stream>>>(lpre, lqkv_dw, bufB, 576, tot);
  }
  // 7. local channel attention
  rownorm_local_kernel<<<BATCH * 384 * 16, 256, 0, stream>>>(bufB, lnorm);
  gram_local_kernel<<<BATCH * HEADS * 16, 576, 0, stream>>>(bufB, lnorm, l_temp, lgram);
  softmax24_kernel<<<(BATCH * HEADS * 16 * 24 + 255) / 256, 256, 0, stream>>>(lgram, BATCH * HEADS * 16 * 24);
  attnv_local_kernel<<<dim3(4096 / 256, BATCH * HEADS * 16), 256, 0, stream>>>(bufB, lgram, local_w, actD);

  // 8. loc = lproj GEMM -> bufC
  gemm_bf16_wmma<<<dim3(N / 128, CCH / 64, BATCH), 256, 0, stream>>>(
      wpool + o_wlproj, actD, bufC, nullptr, CCH, CCH, (long)CCH * N, (long)CCH * N);

  // 9. x1 = x + (1-ld)*glo + ld*loc  (in place into bufC)
  combine_kernel<<<(unsigned)(((long)BATCH * CCH * N) / 256), 256, 0, stream>>>(x, lightd, bufA, bufC);

  // 10. n2 = LN(x1)
  ln_c_kernel<<<(BATCH * N) / 256, 256, 0, stream>>>(bufC, n2_w, n2_b, nrm);

  // 11. ffn_in GEMM -> bufA, dwconv -> bufB, gate -> actD (bf16)
  gemm_bf16_wmma<<<dim3(N / 128, 768 / 64, BATCH), 256, 0, stream>>>(
      wpool + o_wffn1, nrm, bufA, nullptr, 768, 192, (long)CCH * N, (long)768 * N);
  {
    long tot = (long)BATCH * 768 * N;
    dwconv3x3_kernel<<<(unsigned)(tot / 256), 256, 0, stream>>>(bufA, ffn_dw, bufB, 768, tot);
  }
  gate_kernel<<<(unsigned)(((long)BATCH * HID * N) / 256), 256, 0, stream>>>(bufB, actD);

  // 12. out = x1 + ffn_out GEMM (fused residual)
  gemm_bf16_wmma<<<dim3(N / 128, CCH / 64, BATCH), 256, 0, stream>>>(
      wpool + o_wffn2, actD, out, bufC, CCH, HID, (long)HID * N, (long)CCH * N);
}
